// GATLayer_34359738949
// MI455X (gfx1250) — matrix operand order repaired and checked by reading, not compiled, not run on hardware
//
#include <hip/hip_runtime.h>
#include <stdint.h>

#define BATCH 4
#define NNODE 4096
#define FDIM  256
#define NEG_BIG (-9.0e15f)

// LDS row stride in halfs: multiple of 8 so fragment reads are 16B-aligned
// b128 loads; dword stride (stride/2 = 20) keeps store conflicts <= 2-way.
#define LSTR 40

typedef __attribute__((ext_vector_type(16))) _Float16 v16h;
typedef __attribute__((ext_vector_type(8)))  float    v8f;
typedef __attribute__((ext_vector_type(2)))  __fp16   fp16x2;

union Frag16 { v16h v; uint32_t u[8]; uint4 q[2]; };

// f32 x2 -> packed f16x2 in one v_cvt_pk_rtz_f16_f32
__device__ __forceinline__ uint32_t pack2h(float x, float y) {
  union { fp16x2 h; uint32_t u; } t;
  t.h = __builtin_amdgcn_cvt_pkrtz(x, y);
  return t.u;
}

// ---------------------------------------------------------------------------
// Kernel 1: Wh = h @ W  (f32 in, f16 out).  One block = 16 rows x 256 cols,
// 4 waves; wave w owns cols [64w, 64w+64).  W tile staged transposed in LDS
// ([col][k], stride LSTR) so B fragments are two ds_load_b128 per tile.
// ---------------------------------------------------------------------------
__global__ __launch_bounds__(128) void wh_gemm_kernel(
    const float* __restrict__ h, const float* __restrict__ W,
    _Float16* __restrict__ whf16) {
  __shared__ __align__(16) _Float16 sA[16 * LSTR];    // h tile  [row][k]
  __shared__ __align__(16) _Float16 sBT[256 * LSTR];  // W tile  [col][k]

  const int t    = threadIdx.x;
  const int lane = t & 31;
  const int wave = t >> 5;
  const int row0 = blockIdx.x * 16;   // flat row index over B*N

  v8f acc[4];
  #pragma unroll
  for (int tt = 0; tt < 4; ++tt)
    #pragma unroll
    for (int r = 0; r < 8; ++r) acc[tt][r] = 0.0f;

  const int arow   = t >> 3;          // 0..15
  const int akq    = (t & 7) * 4;     // 0..28
  const int kpair  = t & 15;          // two consecutive K rows of W
  const int colseg = (t >> 4) * 32;   // 0..224

  for (int k0 = 0; k0 < FDIM; k0 += 32) {
    __syncthreads();
    // stage A tile (16x32), f32 -> f16, packed pairs
    {
      const float4 v = *(const float4*)&h[(size_t)(row0 + arow) * FDIM + k0 + akq];
      *(uint32_t*)&sA[arow * LSTR + akq + 0] = pack2h(v.x, v.y);
      *(uint32_t*)&sA[arow * LSTR + akq + 2] = pack2h(v.z, v.w);
    }
    // stage B tile transposed: rows k0+2*kpair(+1), cols colseg..+31
    {
      const float* r0 = &W[(size_t)(k0 + 2 * kpair) * FDIM + colseg];
      const float* r1 = r0 + FDIM;
      #pragma unroll
      for (int cc = 0; cc < 32; ++cc)
        *(uint32_t*)&sBT[(colseg + cc) * LSTR + 2 * kpair] = pack2h(r0[cc], r1[cc]);
    }
    __syncthreads();

    // A frag: lanes 0-15 -> row=lane, K {0-7,16-23}; lanes 16-31 -> K {8-15,24-31}
    Frag16 afr;
    const int arowl = lane & 15;
    const int kb    = (lane >> 4) * 8;
    afr.q[0] = *(const uint4*)&sA[arowl * LSTR + kb];
    afr.q[1] = *(const uint4*)&sA[arowl * LSTR + kb + 16];
    // B frags: lanes 0-15 -> K {0-7,16-23}, lanes 16-31 -> K {8-15,24-31}; N = lane&15
    const int kb2 = (lane >> 4) * 8;
    #pragma unroll
    for (int tt = 0; tt < 4; ++tt) {
      const int col = wave * 64 + tt * 16 + (lane & 15);
      Frag16 bfr;
      bfr.q[0] = *(const uint4*)&sBT[col * LSTR + kb2];
      bfr.q[1] = *(const uint4*)&sBT[col * LSTR + kb2 + 16];
      acc[tt] = __builtin_amdgcn_wmma_f32_16x16x32_f16(
          false, afr.v, false, bfr.v, (short)0, acc[tt], false, false);
    }
  }

  // C/D layout: VGPR r -> row = r + 8*(lane>=16), col = base + (lane&15)
  const int hi = lane >> 4;
  #pragma unroll
  for (int tt = 0; tt < 4; ++tt) {
    const int col = wave * 64 + tt * 16 + (lane & 15);
    #pragma unroll
    for (int r = 0; r < 8; ++r)
      whf16[(size_t)(row0 + r + 8 * hi) * FDIM + col] = (_Float16)acc[tt][r];
  }
}

// ---------------------------------------------------------------------------
// Kernel 2: src = Wh @ a[:F], dst = Wh @ a[F:].  One wave per row.
// ---------------------------------------------------------------------------
__global__ __launch_bounds__(128) void src_dst_kernel(
    const _Float16* __restrict__ whf16, const float* __restrict__ a,
    float* __restrict__ src, float* __restrict__ dst) {
  const int lane = threadIdx.x & 31;
  const int wave = threadIdx.x >> 5;
  const int row  = blockIdx.x * 4 + wave;
  float s = 0.f, d = 0.f;
  #pragma unroll
  for (int k = 0; k < 8; ++k) {
    const int c = lane + k * 32;
    const float w = (float)whf16[(size_t)row * FDIM + c];
    s += w * a[c];
    d += w * a[FDIM + c];
  }
  #pragma unroll
  for (int m = 16; m >= 1; m >>= 1) {
    s += __shfl_xor(s, m, 32);
    d += __shfl_xor(d, m, 32);
  }
  if (lane == 0) { src[row] = s; dst[row] = d; }
}

// ---------------------------------------------------------------------------
// Kernel 3: fused masked-softmax attention aggregation (flash style).
// One block = 32 output rows of one batch, 256 threads (8 waves).
// Waves 0-3 own rows 0-15, waves 4-7 rows 16-31; wave w owns cols
// [(w&3)*64, +64).  Single pass over adj (the 268MB roofline term).
// ---------------------------------------------------------------------------
__global__ __launch_bounds__(256) void gat_attn_kernel(
    const int* __restrict__ adj, const _Float16* __restrict__ whf16,
    const float* __restrict__ src, const float* __restrict__ dst,
    float* __restrict__ out) {
  __shared__ __align__(16) _Float16 sP[32 * LSTR];    // P tile (A matrix), f16
  __shared__ __align__(16) _Float16 sBT[256 * LSTR];  // Wh j-tile [col][j], f16
  __shared__ float sScale[32];
  __shared__ float sL[32];

  const int t    = threadIdx.x;
  const int lane = t & 31;
  const int wave = t >> 5;                   // 0..7
  const int b    = blockIdx.x >> 7;          // 128 i-tiles of 32 per batch
  const int i0   = (blockIdx.x & 127) * 32;

  const int irow   = t >> 3;       // row this thread scores (0..31)
  const int jq     = (t & 7) * 4;  // j offset within tile (0..28)
  const int kpair  = t & 15;       // two consecutive j rows of Wh
  const int colseg = (t >> 4) * 16;

  const int rowbase = (wave >> 2) * 16;      // this wave's i-row half
  const int colbase = (wave & 3) * 64;       // this wave's column slab

  const size_t adjRow = ((size_t)b * NNODE + (size_t)(i0 + irow)) * NNODE;
  const float srow = src[b * NNODE + i0 + irow];
  const float* dstb = &dst[b * NNODE];
  const _Float16* whb = &whf16[(size_t)b * NNODE * FDIM];

  float m_run = -__builtin_inff();
  float l_run = 0.f;

  v8f acc[4];
  #pragma unroll
  for (int tt = 0; tt < 4; ++tt)
    #pragma unroll
    for (int r = 0; r < 8; ++r) acc[tt][r] = 0.0f;

  for (int j0 = 0; j0 < NNODE; j0 += 32) {
    __syncthreads();  // protects sP/sBT/sScale reuse from previous iteration

    // --- masked leaky-relu scores for (irow, j0+jq .. +3) ---
    const int4 av = *(const int4*)&adj[adjRow + j0 + jq];
    const float4 dv = *(const float4*)&dstb[j0 + jq];
    float sc[4];
    {
      float e0 = srow + dv.x, e1 = srow + dv.y;
      float e2 = srow + dv.z, e3 = srow + dv.w;
      e0 = (e0 > 0.f) ? e0 : 0.2f * e0;
      e1 = (e1 > 0.f) ? e1 : 0.2f * e1;
      e2 = (e2 > 0.f) ? e2 : 0.2f * e2;
      e3 = (e3 > 0.f) ? e3 : 0.2f * e3;
      sc[0] = (av.x > 0) ? e0 : NEG_BIG;
      sc[1] = (av.y > 0) ? e1 : NEG_BIG;
      sc[2] = (av.z > 0) ? e2 : NEG_BIG;
      sc[3] = (av.w > 0) ? e3 : NEG_BIG;
    }
    // --- online softmax: 8 lanes per row reduce over the 32-j tile ---
    float tmax = fmaxf(fmaxf(sc[0], sc[1]), fmaxf(sc[2], sc[3]));
    #pragma unroll
    for (int m = 1; m <= 4; m <<= 1) tmax = fmaxf(tmax, __shfl_xor(tmax, m, 32));
    const float m_new = fmaxf(m_run, tmax);
    const float p0 = __expf(sc[0] - m_new), p1 = __expf(sc[1] - m_new);
    const float p2 = __expf(sc[2] - m_new), p3 = __expf(sc[3] - m_new);
    float tsum = p0 + p1 + p2 + p3;
    #pragma unroll
    for (int m = 1; m <= 4; m <<= 1) tsum += __shfl_xor(tsum, m, 32);
    const float scale = __expf(m_run - m_new);  // first iter: exp(-inf)=0
    l_run = l_run * scale + tsum;
    m_run = m_new;
    *(uint32_t*)&sP[irow * LSTR + jq + 0] = pack2h(p0, p1);
    *(uint32_t*)&sP[irow * LSTR + jq + 2] = pack2h(p2, p3);
    if ((t & 7) == 0) sScale[irow] = scale;

    // --- stage Wh j-tile transposed via v_perm_b32 pair packing ---
    {
      const uint32_t* r0 =
          (const uint32_t*)&whb[(size_t)(j0 + 2 * kpair) * FDIM + colseg];
      const uint32_t* r1 = r0 + (FDIM / 2);
      #pragma unroll
      for (int cc = 0; cc < 8; ++cc) {
        const uint32_t u0 = r0[cc], u1 = r1[cc];
        *(uint32_t*)&sBT[(colseg + 2 * cc + 0) * LSTR + 2 * kpair] =
            __builtin_amdgcn_perm(u1, u0, 0x05040100u);  // lo halves
        *(uint32_t*)&sBT[(colseg + 2 * cc + 1) * LSTR + 2 * kpair] =
            __builtin_amdgcn_perm(u1, u0, 0x07060302u);  // hi halves
      }
    }
    __syncthreads();

    // --- rescale accumulators by exp(m_old - m_new) per row ---
    const int hi = lane >> 4;
    float rs[8];
    #pragma unroll
    for (int r = 0; r < 8; ++r) rs[r] = sScale[rowbase + r + 8 * hi];
    #pragma unroll
    for (int tt = 0; tt < 4; ++tt)
      #pragma unroll
      for (int r = 0; r < 8; ++r) acc[tt][r] *= rs[r];

    // --- acc += P(16x32) @ Wh_tile(32xN), 4 WMMAs per wave ---
    Frag16 afr;
    const int arowl = rowbase + (lane & 15);
    const int kb    = (lane >> 4) * 8;
    afr.q[0] = *(const uint4*)&sP[arowl * LSTR + kb];
    afr.q[1] = *(const uint4*)&sP[arowl * LSTR + kb + 16];
    const int kb2 = (lane >> 4) * 8;
    #pragma unroll
    for (int tt = 0; tt < 4; ++tt) {
      const int col = colbase + tt * 16 + (lane & 15);
      Frag16 bfr;
      bfr.q[0] = *(const uint4*)&sBT[col * LSTR + kb2];
      bfr.q[1] = *(const uint4*)&sBT[col * LSTR + kb2 + 16];
      acc[tt] = __builtin_amdgcn_wmma_f32_16x16x32_f16(
          false, afr.v, false, bfr.v, (short)0, acc[tt], false, false);
    }
  }

  // --- finalize: divide by softmax denominator, relu, store ---
  __syncthreads();
  if ((t & 7) == 0) sL[irow] = l_run;
  __syncthreads();
  const int hi = lane >> 4;
  float invl[8];
  #pragma unroll
  for (int r = 0; r < 8; ++r) invl[r] = 1.0f / sL[rowbase + r + 8 * hi];
  #pragma unroll
  for (int tt = 0; tt < 4; ++tt) {
    const int col = colbase + tt * 16 + (lane & 15);
    #pragma unroll
    for (int r = 0; r < 8; ++r) {
      const int grow = i0 + rowbase + r + 8 * hi;
      out[((size_t)b * NNODE + grow) * FDIM + col] =
          fmaxf(acc[tt][r] * invl[r], 0.0f);
    }
  }
}

// ---------------------------------------------------------------------------
extern "C" void kernel_launch(void* const* d_in, const int* in_sizes, int n_in,
                              void* d_out, int out_size, void* d_ws, size_t ws_size,
                              hipStream_t stream) {
  const float* h   = (const float*)d_in[0];   // [B,N,F] f32
  const int*   adj = (const int*)d_in[1];     // [B,N,N] i32
  const float* W   = (const float*)d_in[2];   // [F,F]   f32
  const float* a   = (const float*)d_in[3];   // [2F,1]  f32
  float* out = (float*)d_out;                 // [B,N,F] f32

  const size_t BN = (size_t)BATCH * NNODE;
  _Float16* whf16 = (_Float16*)d_ws;                 // BN*F f16 (8 MB)
  float*    src   = (float*)(whf16 + BN * FDIM);     // BN f32
  float*    dst   = src + BN;                        // BN f32

  wh_gemm_kernel<<<(int)(BN / 16), 128, 0, stream>>>(h, W, whf16);
  src_dst_kernel<<<(int)(BN / 4), 128, 0, stream>>>(whf16, a, src, dst);
  gat_attn_kernel<<<(int)(BN / 32), 256, 0, stream>>>(adj, whf16, src, dst, out);
}
